// CoreAttention_69758858821667
// MI455X (gfx1250) — matrix operand order repaired and checked by reading, not compiled, not run on hardware
//
#include <hip/hip_runtime.h>
#include <hip/hip_bf16.h>

typedef __attribute__((ext_vector_type(16))) _Float16 v16h;
typedef __attribute__((ext_vector_type(8)))  _Float16 v8h;
typedef __attribute__((ext_vector_type(4)))  _Float16 v4h;
typedef __attribute__((ext_vector_type(2)))  _Float16 v2h;
typedef __attribute__((ext_vector_type(8)))  float    v8f;

#define BB 4
#define SS 2048
#define HH 16
#define DD 64
#define WIN 512
#define SOFTCAP 50.0f
#define LOG2E 1.4426950408889634f
#define MASKVAL (-1.0e30f)

// ---- DPP16-based 16-lane row max (no LDS traffic) ----
template <int CTRL>
__device__ __forceinline__ float dpp_movf(float x) {
    return __int_as_float(
        __builtin_amdgcn_update_dpp(0, __float_as_int(x), CTRL, 0xF, 0xF, true));
}
__device__ __forceinline__ float rowmax16(float x) {
    x = fmaxf(x, dpp_movf<0x0B1>(x));   // quad_perm xor1
    x = fmaxf(x, dpp_movf<0x04E>(x));   // quad_perm xor2
    x = fmaxf(x, dpp_movf<0x141>(x));   // row_half_mirror (^7)
    x = fmaxf(x, dpp_movf<0x140>(x));   // row_mirror (^15)
    return x;
}

// Single-instruction hardware tanh (CDNA5 V_TANH_F32).
__device__ __forceinline__ float hw_tanh(float x) {
#if __has_builtin(__builtin_amdgcn_tanh_f32)
    return __builtin_amdgcn_tanh_f32(x);
#else
    float y;
    asm("v_tanh_f32 %0, %1\n\t"
        "s_delay_alu instid0(TRANS32_DEP_1)"
        : "=v"(y) : "v"(x));
    return y;
#endif
}

// Raw base-2 exp (v_exp_f32) — softmax runs in the log2 domain.
__device__ __forceinline__ float exp2_raw(float x) {
#if __has_builtin(__builtin_amdgcn_exp2f)
    return __builtin_amdgcn_exp2f(x);
#else
    return exp2f(x);
#endif
}
// Raw v_rcp_f32 for the softmax normalizer.
__device__ __forceinline__ float rcp_raw(float x) {
#if __has_builtin(__builtin_amdgcn_rcpf)
    return __builtin_amdgcn_rcpf(x);
#else
    return 1.0f / x;
#endif
}

// v_cvt_pk_f16_f32 wrapper (builtin returns an __fp16 vector; bit-cast it).
__device__ __forceinline__ v2h pkrtz(float x, float y) {
    return __builtin_bit_cast(v2h, __builtin_amdgcn_cvt_pkrtz(x, y));
}
__device__ __forceinline__ v4h cvt4(float x, float y, float z, float w) {
    const v2h a = pkrtz(x, y);
    const v2h b = pkrtz(z, w);
    return (v4h){a[0], a[1], b[0], b[1]};
}

// Flash attention, sliding-window causal, tanh softcap.
// Block = 128 threads (4 waves). Each block: one (b,h), 64 query rows.
// Each wave: 16 query rows. Key tiles of 32 per iteration,
// register-staged double buffer for global->LDS.
// Row sums computed on the matrix engine via a ones-matrix WMMA.
// Softmax state kept in log2 domain so v_exp_f32 is used raw.
__global__ __launch_bounds__(128)
void fa_swa_softcap_kernel(const float* __restrict__ Qg,
                           const float* __restrict__ Kg,
                           const float* __restrict__ Vg,
                           float* __restrict__ Og) {
    __shared__ _Float16 sK[32 * 64];      // [key][d]  f16
    __shared__ _Float16 sVt[64 * 32];     // [d][key]  f16 (transposed V)
    __shared__ _Float16 sP[4][16 * 32];   // per-wave probs [m][k]

    const int tid  = threadIdx.x;
    const int wave = tid >> 5;
    const int lane = tid & 31;
    const int hi   = (lane >> 4) & 1;   // 0: lanes 0-15, 1: lanes 16-31
    const int ln   = lane & 15;

    const int q0 = blockIdx.x * 64;     // query tile base
    const int h  = blockIdx.y;
    const int b  = blockIdx.z;

    // ---- Load Q A-fragments (16x64 -> two 16x32 f16 A-frags) ----
    const int mrow = q0 + wave * 16 + ln;
    const float* qrow = Qg + ((size_t)((size_t)b * SS + mrow) * HH + h) * DD;
    v16h aQ[2];
#pragma unroll
    for (int f = 0; f < 2; ++f) {
#pragma unroll
        for (int c = 0; c < 2; ++c) {
            const int dbase = f * 32 + hi * 16 + c * 8;
#pragma unroll
            for (int j = 0; j < 8; j += 2) {
                const v2h p = pkrtz(qrow[dbase + j], qrow[dbase + j + 1]);
                aQ[f][c * 8 + j]     = p[0];
                aQ[f][c * 8 + j + 1] = p[1];
            }
        }
    }

    // ---- Online softmax state: o[0..3] = O accum, o[4] = row-sum accum ----
    v8f o[5];
#pragma unroll
    for (int c = 0; c < 5; ++c) o[c] = (v8f){0.f,0.f,0.f,0.f,0.f,0.f,0.f,0.f};
    float mrun[8];                       // running max, log2 domain
#pragma unroll
    for (int r = 0; r < 8; ++r) mrun[r] = MASKVAL;

    // Ones B-matrix: l += P x 1 puts row sums in every column of o[4].
    v16h bOnes;
#pragma unroll
    for (int j = 0; j < 16; ++j) bOnes[j] = (_Float16)1.0f;

    // Key range: j in [q0-511, q0+63], clamped to >= 0, aligned down to 32.
    int klo = q0 - (WIN - 1);
    if (klo < 0) klo = 0;
    const int kb0 = klo & ~31;
    const int kend = q0 + 63;
    const size_t KSTEP = (size_t)32 * HH * DD;   // floats per 32-key tile

    // ---- Per-thread load slots (hoisted addressing) ----
    int keyT[4], ddT[4];
    const float* kptr[4];
    const float* vptr[4];
#pragma unroll
    for (int t = 0; t < 4; ++t) {
        const int lin = t * 128 + tid;          // float4 index in 32x64 tile
        keyT[t] = lin >> 4;                     // 16 float4 per key row
        ddT[t]  = (lin & 15) * 4;
        const size_t off =
            ((size_t)((size_t)b * SS + (kb0 + keyT[t])) * HH + h) * DD + ddT[t];
        kptr[t] = Kg + off;
        vptr[t] = Vg + off;
    }

    // ---- Prologue: stage first tile into registers ----
    float4 stK[4], stV[4];
#pragma unroll
    for (int t = 0; t < 4; ++t) {
        stK[t] = *(const float4*)kptr[t];
        stV[t] = *(const float4*)vptr[t];
        kptr[t] += KSTEP;
        vptr[t] += KSTEP;
    }

    for (int kb = kb0; kb <= kend; kb += 32) {
        __syncthreads();   // previous iteration's LDS consumers done

        // ---- Commit staged registers to LDS (f32 -> f16) ----
#pragma unroll
        for (int t = 0; t < 4; ++t) {
            *(v4h*)&sK[keyT[t] * 64 + ddT[t]] =
                cvt4(stK[t].x, stK[t].y, stK[t].z, stK[t].w);
            sVt[(ddT[t] + 0) * 32 + keyT[t]] = (_Float16)stV[t].x;
            sVt[(ddT[t] + 1) * 32 + keyT[t]] = (_Float16)stV[t].y;
            sVt[(ddT[t] + 2) * 32 + keyT[t]] = (_Float16)stV[t].z;
            sVt[(ddT[t] + 3) * 32 + keyT[t]] = (_Float16)stV[t].w;
        }
        __syncthreads();

        // ---- Issue next tile's global loads early (overlap with compute) ----
        if (kb + 32 <= kend) {
#pragma unroll
            for (int t = 0; t < 4; ++t) {
                stK[t] = *(const float4*)kptr[t];
                stV[t] = *(const float4*)vptr[t];
                __builtin_prefetch(kptr[t] + KSTEP, 0, 1);
                __builtin_prefetch(vptr[t] + KSTEP, 0, 1);
                kptr[t] += KSTEP;
                vptr[t] += KSTEP;
            }
        }

        // ---- Scores: S = Q K^T for 16 rows x 32 keys (two 16x16 C-frags) ----
        v8f c0 = (v8f){0.f,0.f,0.f,0.f,0.f,0.f,0.f,0.f};
        v8f c1 = c0;
        {
            const v16h b00 = *(const v16h*)&sK[(0 * 16 + ln) * 64 +  0 + hi * 16];
            const v16h b01 = *(const v16h*)&sK[(0 * 16 + ln) * 64 + 32 + hi * 16];
            const v16h b10 = *(const v16h*)&sK[(1 * 16 + ln) * 64 +  0 + hi * 16];
            const v16h b11 = *(const v16h*)&sK[(1 * 16 + ln) * 64 + 32 + hi * 16];
            c0 = __builtin_amdgcn_wmma_f32_16x16x32_f16(false, aQ[0], false, b00, (short)0, c0, false, false);
            c0 = __builtin_amdgcn_wmma_f32_16x16x32_f16(false, aQ[1], false, b01, (short)0, c0, false, false);
            c1 = __builtin_amdgcn_wmma_f32_16x16x32_f16(false, aQ[0], false, b10, (short)0, c1, false, false);
            c1 = __builtin_amdgcn_wmma_f32_16x16x32_f16(false, aQ[1], false, b11, (short)0, c1, false, false);
        }

        // ---- Scale, softcap (HW tanh), mask — all in log2 domain ----
        // sc = (SOFTCAP*log2e) * tanh(qk/(8*SOFTCAP)); valid <=> 0 <= i-j < WIN.
        const int d0 = (q0 + wave * 16 + hi * 8) - (kb + ln);   // i-j for r=0,t=0
        float sc0[8], sc1[8];
#pragma unroll
        for (int r = 0; r < 8; ++r) {
            const float s0 = (SOFTCAP * LOG2E) * hw_tanh(c0[r] * (0.125f / SOFTCAP));
            sc0[r] = ((unsigned)(d0 + r) < WIN) ? s0 : MASKVAL;
            const float s1 = (SOFTCAP * LOG2E) * hw_tanh(c1[r] * (0.125f / SOFTCAP));
            sc1[r] = ((unsigned)(d0 - 16 + r) < WIN) ? s1 : MASKVAL;
        }

        // ---- Online softmax update (DPP row max; sums go to the matrix unit) ----
#pragma unroll
        for (int r = 0; r < 8; ++r) {
            const float tm  = rowmax16(fmaxf(sc0[r], sc1[r]));
            const float nm  = fmaxf(mrun[r], tm);
            const float sca = exp2_raw(mrun[r] - nm);
            mrun[r] = nm;

            const float p0 = exp2_raw(sc0[r] - nm);
            const float p1 = exp2_raw(sc1[r] - nm);

            o[0][r] *= sca; o[1][r] *= sca; o[2][r] *= sca;
            o[3][r] *= sca; o[4][r] *= sca;

            // C-layout -> per-wave LDS P buffer [m][k32]
            sP[wave][(r + hi * 8) * 32 +      ln] = (_Float16)p0;
            sP[wave][(r + hi * 8) * 32 + 16 + ln] = (_Float16)p1;
        }

        // Wave-internal LDS RAW: ensure stores landed before A-frag reads.
        asm volatile("s_wait_dscnt 0x0" ::: "memory");

        // ---- Rebuild P as A-fragment (16x32 f16) ----
        v16h aP;
        {
            const v8h pl = *(const v8h*)&sP[wave][ln * 32 + hi * 16];
            const v8h ph = *(const v8h*)&sP[wave][ln * 32 + hi * 16 + 8];
#pragma unroll
            for (int j = 0; j < 8; ++j) { aP[j] = pl[j]; aP[8 + j] = ph[j]; }
        }

        // ---- O += P V ; l += P x 1  (K = 32 keys) ----
#pragma unroll
        for (int c = 0; c < 4; ++c) {
            const v16h bV = *(const v16h*)&sVt[(c * 16 + ln) * 32 + hi * 16];
            o[c] = __builtin_amdgcn_wmma_f32_16x16x32_f16(false, aP, false, bV, (short)0, o[c], false, false);
        }
        o[4] = __builtin_amdgcn_wmma_f32_16x16x32_f16(false, aP, false, bOnes, (short)0, o[4], false, false);
    }

    // ---- Epilogue: normalize and store fp32 [B,S,H,D] ----
#pragma unroll
    for (int r = 0; r < 8; ++r) {
        const int i = q0 + wave * 16 + r + hi * 8;
        const float inv = rcp_raw(o[4][r]);      // row sum (replicated per column)
        float* orow = Og + ((size_t)((size_t)b * SS + i) * HH + h) * DD + ln;
#pragma unroll
        for (int c = 0; c < 4; ++c)
            orow[c * 16] = o[c][r] * inv;
    }
}

extern "C" void kernel_launch(void* const* d_in, const int* in_sizes, int n_in,
                              void* d_out, int out_size, void* d_ws, size_t ws_size,
                              hipStream_t stream) {
    const float* Q = (const float*)d_in[0];
    const float* K = (const float*)d_in[1];
    const float* V = (const float*)d_in[2];
    float* O = (float*)d_out;
    (void)in_sizes; (void)n_in; (void)out_size; (void)d_ws; (void)ws_size;

    dim3 grid(SS / 64, HH, BB);   // (32, 16, 4)
    dim3 block(128);
    fa_swa_softcap_kernel<<<grid, block, 0, stream>>>(Q, K, V, O);
}
